// BI_Interaction_49873160241844
// MI455X (gfx1250) — compile-verified, matrix operand order checked by reading, not run on hardware
//
#include <hip/hip_runtime.h>
#include <hip/hip_bf16.h>

// ---------------------------------------------------------------------------
// MI455X (gfx1250) PaiNN-style interaction block.
//
// Roofline: Wij ([320000,384] f32 = 491 MB) is streamed exactly once ->
// ~23 us floor at 23.3 TB/s. X (15 MB) / V (15 MB) gather tables live in the
// 192 MB L2. All GEMMs together are ~4.3 GFLOP -> noise on WMMA.
//
// Round-2 changes:
//  * Weight matrices are pre-packed once into WMMA B-fragment order (f16,
//    32 B per lane per k-step) -> B fetch = 2x global_load_b128, no addr math.
//  * Edge kernel streams Wij via async global->LDS copies (ASYNCcnt path):
//    per-wave double-buffered LDS staging, 12x global_load_async_to_lds_b128
//    per 4-edge chunk, explicit s_wait_asynccnt; gathers of X/V stay direct.
// ---------------------------------------------------------------------------

typedef __attribute__((ext_vector_type(16))) _Float16 v16h;
typedef __attribute__((ext_vector_type(8)))  float    v8f;

#define F_DIM 128

// ---------------------------------------------------------------------------
// Pack B[128,N] (f32, row-major) into WMMA fragment order as f16:
// Bp[(tile*4 + kk4)*32 + lane] = v16h fragment for column tile `tile`,
// K-step kk4 (K = kk4*32..+31), lane `lane`. ISA 7.12.2 16-bit layout:
// lane holds column (lane&15); group g=(lane>>4): K = {g*8..g*8+7} U
// {16+g*8..16+g*8+7} inside the 32-wide step.
// ---------------------------------------------------------------------------
__global__ void pack_weights(const float* __restrict__ B,
                             v16h* __restrict__ Bp, int N) {
  const int tid   = blockIdx.x * blockDim.x + threadIdx.x;
  const int total = (N / 16) * 4 * 32;
  if (tid >= total) return;
  const int lane = tid & 31;
  const int kk4  = (tid >> 5) & 3;
  const int tile = tid >> 7;
  const int col  = tile * 16 + (lane & 15);
  const int g8   = (lane >> 4) * 8;
  const int kk   = kk4 * 32;
  v16h b;
#pragma unroll
  for (int j = 0; j < 16; ++j) {
    const int k = kk + g8 + ((j < 8) ? j : (8 + j));
    b[j] = (_Float16)B[(size_t)k * N + col];
  }
  Bp[tid] = b;
}

// ---------------------------------------------------------------------------
// C[M,N] = act(A[M,128] @ B + bias), B pre-packed as above.
// 256 threads = 8 waves; wave w owns the 16x16 tile at column
// blockIdx.y*128 + 16*w. M % 16 == 0, N % 128 == 0 -> no masking, EXEC all-1s.
// ---------------------------------------------------------------------------
__global__ void gemm_wmma_k128(const float* __restrict__ A,
                               const v16h* __restrict__ Bp,
                               const float* __restrict__ bias,
                               float* __restrict__ C,
                               int N, int act /*0=none,1=silu*/) {
  const int lane    = threadIdx.x & 31;
  const int wave    = threadIdx.x >> 5;
  const int rowBase = blockIdx.x * 16;
  const int colBase = blockIdx.y * 128 + wave * 16;
  const int row     = rowBase + (lane & 15);
  const int col     = colBase + (lane & 15);
  const int g8      = (lane >> 4) * 8;
  const v16h* bp    = Bp + (size_t)(colBase >> 4) * 128 + lane;

  v8f acc = {};
#pragma unroll
  for (int kk4 = 0; kk4 < 4; ++kk4) {
    const int kk = kk4 * 32;
    v16h a;
#pragma unroll
    for (int j = 0; j < 16; ++j) {
      const int k = kk + g8 + ((j < 8) ? j : (8 + j));
      a[j] = (_Float16)A[(size_t)row * 128 + k];
    }
    const v16h b = bp[kk4 * 32];               // 2x global_load_b128
    acc = __builtin_amdgcn_wmma_f32_16x16x32_f16(
        false, a, false, b, (short)0, acc, false, false);
  }

  const float bia = bias ? bias[col] : 0.0f;
#pragma unroll
  for (int r = 0; r < 8; ++r) {
    const int m = rowBase + (lane >> 4) * 8 + r;
    float x = acc[r] + bia;
    if (act == 1) x = x / (1.0f + __expf(-x));           // silu
    C[(size_t)m * N + col] = x;
  }
}

// ---------------------------------------------------------------------------
// ctx[n,c] = sum_a vmix[n,a,c] * vmix[n,a,128+c]   (vmix: [3N,256])
// ---------------------------------------------------------------------------
__global__ void ctx_kernel(const float* __restrict__ vmix,
                           float* __restrict__ ctx, int nAtoms) {
  const int idx = blockIdx.x * blockDim.x + threadIdx.x;
  if (idx >= nAtoms * F_DIM) return;
  const int n = idx >> 7, c = idx & 127;
  const float* r = vmix + (size_t)n * 3 * 256;
  float acc = 0.0f;
#pragma unroll
  for (int a = 0; a < 3; ++a)
    acc += r[a * 256 + c] * r[a * 256 + 128 + c];
  ctx[idx] = acc;
}

__global__ void zero_kernel(float* __restrict__ p, int count) {
  for (int i = blockIdx.x * blockDim.x + threadIdx.x; i < count;
       i += gridDim.x * blockDim.x)
    p[i] = 0.0f;
}

// ---------------------------------------------------------------------------
// Edge scatter kernel with async global->LDS streaming of Wij.
// Each wave owns EK_EDGES_PER_WAVE consecutive edges and a private
// double-buffered LDS staging area (no cross-wave sync). Lane owns channels
// [lane*4, lane*4+4). Sorted idx_i -> register accumulation per run, flushed
// with global_atomic_add_f32.
// ---------------------------------------------------------------------------
#define EK_WAVES          8      // waves per block
#define EK_CHUNK          4      // edges staged per async chunk
#define EK_EDGES_PER_WAVE 256
#define EK_LOADS_PER_CHUNK 12    // EK_CHUNK*384 floats / (32 lanes * 4 floats)

__global__ void edge_kernel(const float* __restrict__ X,
                            const float* __restrict__ Wij,
                            const float* __restrict__ V,
                            const float* __restrict__ dir,
                            const int* __restrict__ idx_i,
                            const int* __restrict__ idx_j,
                            float* __restrict__ dh,
                            float* __restrict__ dv, int nEdges) {
  __shared__ __align__(16) float ebuf[EK_WAVES][2][EK_CHUNK * 384];

  const int lane = threadIdx.x & 31;
  const int wave = threadIdx.x >> 5;
  const int c4   = lane * 4;

  const int e0 = (blockIdx.x * EK_WAVES + wave) * EK_EDGES_PER_WAVE;
  if (e0 >= nEdges) return;                         // wave-uniform exit
  const int e1 = (e0 + EK_EDGES_PER_WAVE < nEdges) ? e0 + EK_EDGES_PER_WAVE
                                                   : nEdges;

  const unsigned lbase[2] = {
      (unsigned)(size_t)&ebuf[wave][0][0],
      (unsigned)(size_t)&ebuf[wave][1][0]};

  // Issue one full chunk [ce, ce+EK_CHUNK) into buffer b (async, 12 ops).
  auto issue = [&](int ce, int b) {
    const float* g = Wij + (size_t)ce * 384;
#pragma unroll
    for (int k = 0; k < EK_LOADS_PER_CHUNK; ++k) {
      const int off = (k * 32 + lane) * 4;          // float index in chunk
      asm volatile("global_load_async_to_lds_b128 %0, %1, off"
                   :: "v"(lbase[b] + (unsigned)off * 4u), "v"(g + off)
                   : "memory");
    }
  };

  float accDh0 = 0.f, accDh1 = 0.f, accDh2 = 0.f, accDh3 = 0.f;
  float accDv[3][4] = {};
  int cur = -1;

  auto flush = [&]() {
    if (cur >= 0) {
      const size_t hb = (size_t)cur * 128 + c4;
      atomicAdd(&dh[hb + 0], accDh0);
      atomicAdd(&dh[hb + 1], accDh1);
      atomicAdd(&dh[hb + 2], accDh2);
      atomicAdd(&dh[hb + 3], accDh3);
#pragma unroll
      for (int a = 0; a < 3; ++a) {
        const size_t vbf = (size_t)cur * 384 + a * 128 + c4;
#pragma unroll
        for (int u = 0; u < 4; ++u) atomicAdd(&dv[vbf + u], accDv[a][u]);
      }
    }
    accDh0 = accDh1 = accDh2 = accDh3 = 0.f;
#pragma unroll
    for (int a = 0; a < 3; ++a)
#pragma unroll
      for (int u = 0; u < 4; ++u) accDv[a][u] = 0.f;
  };

  bool curAsync = (e0 + EK_CHUNK <= e1);
  if (curAsync) issue(e0, 0);
  int parity = 0;

  for (int ce = e0; ce < e1; ce += EK_CHUNK) {
    const int cnt = ((e1 - ce) < EK_CHUNK) ? (e1 - ce) : EK_CHUNK;
    const bool nxtAsync = (ce + 2 * EK_CHUNK <= e1);
    if (nxtAsync) issue(ce + EK_CHUNK, parity ^ 1);
    if (curAsync) {
      if (nxtAsync) asm volatile("s_wait_asynccnt 0xc" ::: "memory");
      else          asm volatile("s_wait_asynccnt 0x0" ::: "memory");
    }
    __builtin_prefetch(&dir[(size_t)(ce + 2 * EK_CHUNK) * 3], 0, 1);

    for (int ee = 0; ee < cnt; ++ee) {
      const int e = ce + ee;
      const int i = idx_i[e];
      if (i != cur) { flush(); cur = i; }
      const int j = idx_j[e];

      float4 wA, wB, wC;
      if (curAsync) {
        const float* lb = &ebuf[wave][parity][ee * 384];
        wA = *(const float4*)(lb + c4);             // ds_load_b128
        wB = *(const float4*)(lb + 128 + c4);
        wC = *(const float4*)(lb + 256 + c4);
      } else {                                      // tail fallback
        const float* gb = Wij + (size_t)e * 384;
        wA = *(const float4*)(gb + c4);
        wB = *(const float4*)(gb + 128 + c4);
        wC = *(const float4*)(gb + 256 + c4);
      }

      const size_t xj = (size_t)j * 384;
      const float4 xA = *(const float4*)(X + xj + c4);
      const float4 xB = *(const float4*)(X + xj + 128 + c4);
      const float4 xC = *(const float4*)(X + xj + 256 + c4);
      const float d0 = dir[(size_t)e * 3 + 0];
      const float d1 = dir[(size_t)e * 3 + 1];
      const float d2 = dir[(size_t)e * 3 + 2];
      const size_t vj = (size_t)j * 384;
      const float4 V0 = *(const float4*)(V + vj + c4);
      const float4 V1 = *(const float4*)(V + vj + 128 + c4);
      const float4 V2 = *(const float4*)(V + vj + 256 + c4);

      accDh0 += wA.x * xA.x; accDh1 += wA.y * xA.y;
      accDh2 += wA.z * xA.z; accDh3 += wA.w * xA.w;

      const float4 dvR = {wB.x * xB.x, wB.y * xB.y, wB.z * xB.z, wB.w * xB.w};
      const float4 dvv = {wC.x * xC.x, wC.y * xC.y, wC.z * xC.z, wC.w * xC.w};

      accDv[0][0] += dvR.x * d0 + dvv.x * V0.x;
      accDv[0][1] += dvR.y * d0 + dvv.y * V0.y;
      accDv[0][2] += dvR.z * d0 + dvv.z * V0.z;
      accDv[0][3] += dvR.w * d0 + dvv.w * V0.w;
      accDv[1][0] += dvR.x * d1 + dvv.x * V1.x;
      accDv[1][1] += dvR.y * d1 + dvv.y * V1.y;
      accDv[1][2] += dvR.z * d1 + dvv.z * V1.z;
      accDv[1][3] += dvR.w * d1 + dvv.w * V1.w;
      accDv[2][0] += dvR.x * d2 + dvv.x * V2.x;
      accDv[2][1] += dvR.y * d2 + dvv.y * V2.y;
      accDv[2][2] += dvR.z * d2 + dvv.z * V2.z;
      accDv[2][3] += dvR.w * d2 + dvv.w * V2.w;
    }
    curAsync = nxtAsync;
    parity ^= 1;
  }
  flush();
}

// ---------------------------------------------------------------------------
// Fused epilogue: one block (128 threads) per atom.
// ---------------------------------------------------------------------------
__device__ __forceinline__ float block_reduce_sum(float val, float* sbuf) {
  const int t = threadIdx.x;
  sbuf[t] = val;
  __syncthreads();
#pragma unroll
  for (int s = 64; s > 0; s >>= 1) {
    if (t < s) sbuf[t] += sbuf[t + s];
    __syncthreads();
  }
  const float r = sbuf[0];
  __syncthreads();
  return r;
}

__global__ void final_kernel(const float* __restrict__ h,
                             const float* __restrict__ v,
                             const float* __restrict__ dh,
                             const float* __restrict__ dv,
                             const float* __restrict__ ctx,
                             const float* __restrict__ R,
                             const float* __restrict__ Fg,
                             const float* __restrict__ Wv,
                             const float* __restrict__ gamma,
                             const float* __restrict__ beta,
                             float* __restrict__ q,
                             float* __restrict__ mu) {
  __shared__ float sbuf[128];
  const int n = blockIdx.x;
  const int c = threadIdx.x;
  const size_t nc = (size_t)n * 128 + c;

  const float dhv = dh[nc];
  const float s = block_reduce_sum(dhv * Wv[c], sbuf);   // dh @ Wv_remain

  const float qpre = h[nc] + R[nc] + Fg[nc] * ctx[nc];
  const float mean = block_reduce_sum(qpre, sbuf) * (1.0f / 128.0f);
  const float d    = qpre - mean;
  const float var  = block_reduce_sum(d * d, sbuf) * (1.0f / 128.0f);
  q[nc] = d * rsqrtf(var + 1e-5f) * gamma[c] + beta[c];

  const size_t vb = (size_t)n * 384 + c;
  const float x0 = v[vb + 0 * 128] * (1.0f + s) + dv[vb + 0 * 128];
  const float x1 = v[vb + 1 * 128] * (1.0f + s) + dv[vb + 1 * 128];
  const float x2 = v[vb + 2 * 128] * (1.0f + s) + dv[vb + 2 * 128];
  const float sq   = x0 * x0 + x1 * x1 + x2 * x2;
  const float msq  = block_reduce_sum(sq, sbuf) * (1.0f / 128.0f);
  const float rinv = rsqrtf(msq + 1e-8f);
  mu[vb + 0 * 128] = x0 * rinv;
  mu[vb + 1 * 128] = x1 * rinv;
  mu[vb + 2 * 128] = x2 * rinv;
}

// ---------------------------------------------------------------------------
// Host launch.
// ---------------------------------------------------------------------------
extern "C" void kernel_launch(void* const* d_in, const int* in_sizes, int n_in,
                              void* d_out, int out_size, void* d_ws, size_t ws_size,
                              hipStream_t stream) {
  const float* h     = (const float*)d_in[0];
  const float* v     = (const float*)d_in[1];
  const float* H     = (const float*)d_in[2];
  const float* V     = (const float*)d_in[3];
  const float* Wij   = (const float*)d_in[4];
  const float* dir   = (const float*)d_in[5];
  const float* W1    = (const float*)d_in[6];
  const float* b1    = (const float*)d_in[7];
  const float* W2    = (const float*)d_in[8];
  const float* b2    = (const float*)d_in[9];
  const float* Wmix  = (const float*)d_in[10];
  const float* WhR   = (const float*)d_in[11];
  const float* WhF   = (const float*)d_in[12];
  const float* WvR   = (const float*)d_in[13];
  const float* gamma = (const float*)d_in[14];
  const float* beta  = (const float*)d_in[15];
  const int*   idx_i = (const int*)d_in[16];
  const int*   idx_j = (const int*)d_in[17];

  const int N = in_sizes[0] / F_DIM;   // 10000 (multiple of 16)
  const int E = in_sizes[16];          // 320000

  // Workspace layout (floats):
  float* ws = (float*)d_ws;
  size_t off = 0;
  float* dh   = ws + off; off += (size_t)N * 128;
  float* dv   = ws + off; off += (size_t)N * 384;
  float* X    = ws + off; off += (size_t)N * 384;
  float* T    = ws + off; off += (size_t)N * 128;
  float* vmix = ws + off; off += (size_t)N * 768;     // [3N,256]; reused below
  float* ctx  = ws + off; off += (size_t)N * 128;
  float* R    = vmix;                                 // reuse after ctx
  float* Fg   = vmix + (size_t)N * 128;
  // Packed f16 weights (v16h = 8 floats each):
  v16h* pW1   = (v16h*)(ws + off); off += (size_t)(128 / 16) * 128 * 8;
  v16h* pW2   = (v16h*)(ws + off); off += (size_t)(384 / 16) * 128 * 8;
  v16h* pWmix = (v16h*)(ws + off); off += (size_t)(256 / 16) * 128 * 8;
  v16h* pWhR  = (v16h*)(ws + off); off += (size_t)(128 / 16) * 128 * 8;
  v16h* pWhF  = (v16h*)(ws + off); off += (size_t)(128 / 16) * 128 * 8;

  // Pack all weight matrices into WMMA fragment order (tiny, one-shot).
  pack_weights<<<(8  * 128 + 255) / 256, 256, 0, stream>>>(W1,   pW1,   128);
  pack_weights<<<(24 * 128 + 255) / 256, 256, 0, stream>>>(W2,   pW2,   384);
  pack_weights<<<(16 * 128 + 255) / 256, 256, 0, stream>>>(Wmix, pWmix, 256);
  pack_weights<<<(8  * 128 + 255) / 256, 256, 0, stream>>>(WhR,  pWhR,  128);
  pack_weights<<<(8  * 128 + 255) / 256, 256, 0, stream>>>(WhF,  pWhF,  128);

  // Zero scatter targets (dh, dv contiguous).
  zero_kernel<<<512, 256, 0, stream>>>(dh, N * 128 + N * 384);

  const dim3 blk256(256);
  // T = silu(H @ W1 + b1)            [N,128]
  gemm_wmma_k128<<<dim3(N / 16, 1), blk256, 0, stream>>>(H, pW1, b1, T, 128, 1);
  // X = T @ W2 + b2                  [N,384]
  gemm_wmma_k128<<<dim3(N / 16, 3), blk256, 0, stream>>>(T, pW2, b2, X, 384, 0);
  // vmix = v @ Wmix                  [3N,256]
  gemm_wmma_k128<<<dim3(3 * N / 16, 2), blk256, 0, stream>>>(v, pWmix, nullptr, vmix, 256, 0);
  // ctx = sum_a vV*vW                [N,128]
  ctx_kernel<<<(N * 128 + 255) / 256, 256, 0, stream>>>(vmix, ctx, N);

  // Edge stream (async Wij -> LDS, sorted-segment scatter).
  const int totWaves = (E + EK_EDGES_PER_WAVE - 1) / EK_EDGES_PER_WAVE;
  const int eBlocks  = (totWaves + EK_WAVES - 1) / EK_WAVES;
  edge_kernel<<<eBlocks, EK_WAVES * 32, 0, stream>>>(
      X, Wij, V, dir, idx_i, idx_j, dh, dv, E);

  // R = dh @ Wh_remain ; Fg = dh @ Wh_forget      [N,128] each
  gemm_wmma_k128<<<dim3(N / 16, 1), blk256, 0, stream>>>(dh, pWhR, nullptr, R, 128, 0);
  gemm_wmma_k128<<<dim3(N / 16, 1), blk256, 0, stream>>>(dh, pWhF, nullptr, Fg, 128, 0);

  // Fused LN + vector-RMS epilogue -> q, mu
  float* q  = (float*)d_out;
  float* mu = q + (size_t)N * 128;
  final_kernel<<<N, 128, 0, stream>>>(h, v, dh, dv, ctx, R, Fg, WvR, gamma, beta, q, mu);
}
